// Decoder_7834020348405
// MI455X (gfx1250) — matrix operand order repaired and checked by reading, not compiled, not run on hardware
//
#include <hip/hip_runtime.h>
#include <hip/hip_bf16.h>

#define BB 128
#define TT 256
#define MM 256
#define PP 256

typedef _Float16 half8 __attribute__((ext_vector_type(8)));
typedef _Float16 v16h  __attribute__((ext_vector_type(16)));
typedef float    v8f   __attribute__((ext_vector_type(8)));

// ---------- WMMA fragment helpers (ISA 7.12.2 layouts, wave32) ----------

// A fragment: 16x32 f16 tile from a row-major matrix. Lane l (l&15)=row,
// K halves: [kb, kb+8) and [kb+16, kb+24), kb = (l>>4)*8. Two 16B loads.
__device__ __forceinline__ v16h load_a_frag(const _Float16* base, int lda, int lane) {
    int r  = lane & 15;
    int kb = (lane >> 4) << 3;
    const _Float16* p = base + r * lda + kb;
    half8 lo = *(const half8*)(p);
    half8 hi = *(const half8*)(p + 16);
    v16h out;
#pragma unroll
    for (int i = 0; i < 8; ++i) { out[i] = lo[i]; out[i + 8] = hi[i]; }
    return out;
}

// B fragment: 32x16 (KxN) tile loaded from an N-major (pre-transposed) weight:
// lane l: N = l&15, K halves = [kb2, kb2+8) and [kb2+16, kb2+24), kb2 = (l>>4)*8. Two 16B loads.
__device__ __forceinline__ v16h load_bt_frag(const _Float16* baseT, int ldb, int lane) {
    int n   = lane & 15;
    int kb2 = (lane >> 4) << 3;
    const _Float16* p = baseT + n * ldb + kb2;
    half8 lo = *(const half8*)(p);
    half8 hi = *(const half8*)(p + 16);
    v16h out;
#pragma unroll
    for (int i = 0; i < 8; ++i) { out[i] = lo[i]; out[i + 8] = hi[i]; }
    return out;
}

__device__ __forceinline__ v8f wmma_f16(v16h a, v16h b, v8f c) {
    return __builtin_amdgcn_wmma_f32_16x16x32_f16(false, a, false, b, (short)0, c, false, false);
}

__device__ __forceinline__ float sigm(float x) { return 1.0f / (1.0f + __expf(-x)); }

// ---------- one-time prep kernels ----------

__global__ void k_f32_to_f16(const float* __restrict__ src, _Float16* __restrict__ dst, int n) {
    int i = blockIdx.x * blockDim.x + threadIdx.x;
    if (i < n) dst[i] = (_Float16)src[i];
}

// src: rows x cols (K x N) f32  ->  dst: cols x rows (N x K) f16
__global__ void k_transpose_f16(const float* __restrict__ src, _Float16* __restrict__ dst,
                                int rows, int cols) {
    int i = blockIdx.x * blockDim.x + threadIdx.x;
    if (i < rows * cols) {
        int k = i / cols, n = i % cols;
        dst[n * rows + k] = (_Float16)src[i];
    }
}

// pre = enc_h @ W1 + b1 : (B*T, M) x (M, M). One 16x16 tile per wave.
__global__ __launch_bounds__(256) void k_pre_gemm(const _Float16* __restrict__ A,
                                                  const _Float16* __restrict__ W1T,
                                                  const float* __restrict__ b1,
                                                  float* __restrict__ out) {
    int lane = threadIdx.x & 31;
    int wave = threadIdx.x >> 5;
    int id = blockIdx.x * 8 + wave;          // tile id over 2048 x 16 tiles
    int rt = id >> 4, ct = id & 15;
    int row0 = rt * 16, col0 = ct * 16;
    v8f acc = {};
#pragma unroll
    for (int k = 0; k < MM; k += 32) {
        v16h a = load_a_frag(A + (size_t)row0 * MM + k, MM, lane);
        v16h b = load_bt_frag(W1T + col0 * MM + k, MM, lane);
        acc = wmma_f16(a, b, acc);
    }
    int n = lane & 15, rb = (lane >> 4) << 3;
    float bias = b1[col0 + n];
#pragma unroll
    for (int r = 0; r < 8; ++r)
        out[(size_t)(row0 + rb + r) * MM + col0 + n] = acc[r] + bias;
}

__global__ void k_init_state(const float* __restrict__ h0, const float* __restrict__ c0,
                             float* __restrict__ h, float* __restrict__ c,
                             _Float16* __restrict__ h16, float* __restrict__ ctx, int n) {
    int i = blockIdx.x * blockDim.x + threadIdx.x;
    if (i < n) {
        float hv = h0[i];
        h[i] = hv; c[i] = c0[i]; h16[i] = (_Float16)hv; ctx[i] = 0.0f;
    }
}

// ---------- per-step kernels ----------

// One block per 16 batch rows (grid = 8), 256 threads = 8 waves.
// Computes xin, all four LSTM gate tiles in-register via WMMA, h/c, then
// q = [h,c] @ W2 + b2 via WMMA with A sourced from LDS.
__global__ __launch_bounds__(256) void k_lstm_step(
    int step,
    const float* __restrict__ data, const float* __restrict__ Wd, const float* __restrict__ bd,
    const float* __restrict__ Wk, const float* __restrict__ bl,
    const _Float16* __restrict__ WrT, const _Float16* __restrict__ W2T,
    const float* __restrict__ b2,
    const float* __restrict__ ctx_prev, const _Float16* __restrict__ h16_in,
    const float* __restrict__ c_prev,
    float* __restrict__ h_out, float* __restrict__ c_out,
    _Float16* __restrict__ h16_out, float* __restrict__ q_out) {

    __shared__ float red[16][17];
    __shared__ float xin_s[16];
    __shared__ __align__(16) _Float16 hc16[16][2 * PP];   // [h | c] per row, f16

    const int tid = threadIdx.x;
    const int b0 = blockIdx.x * 16;

    // ---- xin[r] = x_t*Wd[0] + ctx . Wd[1:] + bd  (16 threads/row reduction)
    {
        int r = tid >> 4, c16 = tid & 15;
        float s = 0.0f;
        for (int m = c16; m < MM; m += 16)
            s += ctx_prev[(b0 + r) * MM + m] * Wd[1 + m];
        red[r][c16] = s;
    }
    __syncthreads();
    if (tid < 16) {
        float s = 0.0f;
#pragma unroll
        for (int c = 0; c < 16; ++c) s += red[tid][c];
        xin_s[tid] = data[(b0 + tid) * (TT - 1) + step] * Wd[0] + s + bd[0];
    }
    __syncthreads();

    const int lane = tid & 31, wave = tid >> 5;
    const int n = lane & 15, rb = (lane >> 4) << 3;

    // A fragments of h_prev (16 rows x K=256), shared by all gate tiles of this wave
    v16h afr[8];
#pragma unroll
    for (int kk = 0; kk < 8; ++kk)
        afr[kk] = load_a_frag(h16_in + b0 * PP + kk * 32, PP, lane);

    // each wave owns 2 p-tiles; computes i,f,g,o tiles for them in-register
    for (int pi = 0; pi < 2; ++pi) {
        int pt = wave * 2 + pi;          // 0..15
        int p0 = pt * 16;
        v8f acc[4] = {};
#pragma unroll
        for (int g = 0; g < 4; ++g) {
            int col0 = g * PP + p0;      // column in 4P gate space
#pragma unroll
            for (int kk = 0; kk < 8; ++kk) {
                v16h b = load_bt_frag(WrT + col0 * PP + kk * 32, PP, lane);
                acc[g] = wmma_f16(afr[kk], b, acc[g]);
            }
        }
#pragma unroll
        for (int e = 0; e < 8; ++e) {
            int r = rb + e;
            int p = p0 + n;
            float xin = xin_s[r];
            float zi = acc[0][e] + xin * Wk[0 * PP + p] + bl[0 * PP + p];
            float zf = acc[1][e] + xin * Wk[1 * PP + p] + bl[1 * PP + p];
            float zg = acc[2][e] + xin * Wk[2 * PP + p] + bl[2 * PP + p];
            float zo = acc[3][e] + xin * Wk[3 * PP + p] + bl[3 * PP + p];
            float cv = sigm(zf) * c_prev[(b0 + r) * PP + p] + sigm(zi) * tanhf(zg);
            float hv = sigm(zo) * tanhf(cv);
            c_out[(b0 + r) * PP + p] = cv;
            h_out[(b0 + r) * PP + p] = hv;
            h16_out[(b0 + r) * PP + p] = (_Float16)hv;
            hc16[r][p] = (_Float16)hv;
            hc16[r][PP + p] = (_Float16)cv;
        }
    }
    __syncthreads();

    // ---- q = [h,c] @ W2 + b2 : (16 x 256), K = 512, A from LDS
    for (int qi = 0; qi < 2; ++qi) {
        int ct = wave * 2 + qi;          // 0..15
        int col0 = ct * 16;
        v8f acc = {};
#pragma unroll
        for (int kk = 0; kk < 16; ++kk) {
            v16h a = load_a_frag(&hc16[0][0] + kk * 32, 2 * PP, lane);
            v16h b = load_bt_frag(W2T + col0 * (2 * PP) + kk * 32, 2 * PP, lane);
            acc = wmma_f16(a, b, acc);
        }
        float bias = b2[col0 + n];
#pragma unroll
        for (int e = 0; e < 8; ++e)
            q_out[(b0 + rb + e) * MM + col0 + n] = acc[e] + bias;
    }
}

// One block per batch row. score -> softmax -> context. pre/enc_h are L2-resident.
__global__ __launch_bounds__(256) void k_attn_step(
    const float* __restrict__ pre, const float* __restrict__ enc_h,
    const float* __restrict__ q, const float* __restrict__ Wv,
    const float* __restrict__ bv, float* __restrict__ ctx) {

    __shared__ float qs[MM], wv_s[MM], sc[TT], tmp[256];
    const int b = blockIdx.x, tid = threadIdx.x;
    qs[tid] = q[b * MM + tid];
    wv_s[tid] = Wv[tid];
    __syncthreads();

    // score for t = tid
    const float4* prow4 = (const float4*)(pre + ((size_t)b * TT + tid) * MM);
    float s = bv[0];
#pragma unroll 4
    for (int m4 = 0; m4 < MM / 4; ++m4) {
        float4 pv = prow4[m4];
        int m = m4 * 4;
        s += tanhf(pv.x + qs[m + 0]) * wv_s[m + 0];
        s += tanhf(pv.y + qs[m + 1]) * wv_s[m + 1];
        s += tanhf(pv.z + qs[m + 2]) * wv_s[m + 2];
        s += tanhf(pv.w + qs[m + 3]) * wv_s[m + 3];
    }

    // softmax over T
    tmp[tid] = s;
    __syncthreads();
    for (int off = 128; off > 0; off >>= 1) {
        if (tid < off) tmp[tid] = fmaxf(tmp[tid], tmp[tid + off]);
        __syncthreads();
    }
    float mx = tmp[0];
    __syncthreads();
    float e = __expf(s - mx);
    tmp[tid] = e;
    __syncthreads();
    for (int off = 128; off > 0; off >>= 1) {
        if (tid < off) tmp[tid] += tmp[tid + off];
        __syncthreads();
    }
    float inv = 1.0f / tmp[0];
    sc[tid] = e * inv;
    __syncthreads();

    // ctx[b, m=tid] = sum_t beta[t] * enc_h[b,t,m]  (coalesced along m)
    float acc = 0.0f;
    const float* eb = enc_h + (size_t)b * TT * MM + tid;
    for (int t = 0; t < TT; ++t) acc += sc[t] * eb[(size_t)t * MM];
    ctx[b * MM + tid] = acc;
}

__global__ void k_final_concat(const float* __restrict__ h, const float* __restrict__ ctx,
                               float* __restrict__ out) {
    int i = blockIdx.x * blockDim.x + threadIdx.x;
    if (i < BB * (PP + MM)) {
        int b = i >> 9;      // / 512
        int j = i & 511;
        out[i] = (j < PP) ? h[b * PP + j] : ctx[b * MM + (j - PP)];
    }
}

// ---------- host driver ----------

extern "C" void kernel_launch(void* const* d_in, const int* in_sizes, int n_in,
                              void* d_out, int out_size, void* d_ws, size_t ws_size,
                              hipStream_t stream) {
    const float* data  = (const float*)d_in[0];   // (B, T-1, 1)
    const float* enc_h = (const float*)d_in[1];   // (B, T, M)
    const float* h0    = (const float*)d_in[2];   // (B, P)
    const float* c0    = (const float*)d_in[3];   // (B, P)
    const float* Wd    = (const float*)d_in[4];   // (1+M, 1)
    const float* bd    = (const float*)d_in[5];   // (1,)
    const float* Wk    = (const float*)d_in[6];   // (1, 4P)
    const float* Wr    = (const float*)d_in[7];   // (P, 4P)
    const float* bl    = (const float*)d_in[8];   // (4P,)
    const float* W1    = (const float*)d_in[9];   // (M, M)
    const float* b1    = (const float*)d_in[10];  // (M,)
    const float* W2    = (const float*)d_in[11];  // (2P, M)
    const float* b2    = (const float*)d_in[12];  // (M,)
    const float* Wv    = (const float*)d_in[13];  // (M, 1)
    const float* bv    = (const float*)d_in[14];  // (1,)

    // workspace bump allocator (256B aligned)
    char* w = (char*)d_ws;
    auto alloc = [&](size_t bytes) -> void* {
        void* p = (void*)w;
        w += (bytes + 255) & ~(size_t)255;
        return p;
    };
    _Float16* enc16 = (_Float16*)alloc((size_t)BB * TT * MM * sizeof(_Float16));
    float*    pre   = (float*)   alloc((size_t)BB * TT * MM * sizeof(float));
    _Float16* W1T   = (_Float16*)alloc((size_t)MM * MM * sizeof(_Float16));
    _Float16* WrT   = (_Float16*)alloc((size_t)4 * PP * PP * sizeof(_Float16));
    _Float16* W2T   = (_Float16*)alloc((size_t)2 * PP * MM * sizeof(_Float16));
    float*    hbuf  = (float*)   alloc((size_t)BB * PP * sizeof(float));
    float*    cbuf  = (float*)   alloc((size_t)BB * PP * sizeof(float));
    _Float16* h16A  = (_Float16*)alloc((size_t)BB * PP * sizeof(_Float16));
    _Float16* h16B  = (_Float16*)alloc((size_t)BB * PP * sizeof(_Float16));
    float*    qbuf  = (float*)   alloc((size_t)BB * MM * sizeof(float));
    float*    ctxb  = (float*)   alloc((size_t)BB * MM * sizeof(float));

    // one-time prep
    k_f32_to_f16<<<(BB * TT * MM + 255) / 256, 256, 0, stream>>>(enc_h, enc16, BB * TT * MM);
    k_transpose_f16<<<(MM * MM + 255) / 256, 256, 0, stream>>>(W1, W1T, MM, MM);
    k_transpose_f16<<<(PP * 4 * PP + 255) / 256, 256, 0, stream>>>(Wr, WrT, PP, 4 * PP);
    k_transpose_f16<<<(2 * PP * MM + 255) / 256, 256, 0, stream>>>(W2, W2T, 2 * PP, MM);
    k_pre_gemm<<<(BB * TT / 16) * (MM / 16) / 8, 256, 0, stream>>>(enc16, W1T, b1, pre);
    k_init_state<<<(BB * PP + 255) / 256, 256, 0, stream>>>(h0, c0, hbuf, cbuf, h16A, ctxb, BB * PP);

    // sequential scan: 255 steps, h16 ping-pong
    for (int s = 0; s < TT - 1; ++s) {
        const _Float16* h16_in  = (s & 1) ? h16B : h16A;
        _Float16*       h16_out = (s & 1) ? h16A : h16B;
        k_lstm_step<<<BB / 16, 256, 0, stream>>>(s, data, Wd, bd, Wk, bl, WrT, W2T, b2,
                                                 ctxb, h16_in, cbuf,
                                                 hbuf, cbuf, h16_out, qbuf);
        k_attn_step<<<BB, 256, 0, stream>>>(pre, enc_h, qbuf, Wv, bv, ctxb);
    }

    k_final_concat<<<(BB * (PP + MM) + 255) / 256, 256, 0, stream>>>(hbuf, ctxb, (float*)d_out);
}
